// GptOssAttention_42820823941419
// MI455X (gfx1250) — compile-verified, matrix operand order checked by reading, not run on hardware
//
#include <hip/hip_runtime.h>
#include <hip/hip_bf16.h>
#include <math.h>

// ---------------- problem constants (from reference) ----------------
#define DMODEL 2880
#define NHEADS 64
#define KVHEADS 8
#define HDIM 64
#define BATCH 2
#define SEQ 1024
#define TTOK (BATCH * SEQ)       // 2048
#define NQ (NHEADS * HDIM)       // 4096
#define NKV (KVHEADS * HDIM)     // 512
#define WINDOW 128
#define NEG_INF -1e30f

typedef __attribute__((ext_vector_type(16))) __bf16 v16bf;
typedef __attribute__((ext_vector_type(8)))  __bf16 v8bf;
typedef __attribute__((ext_vector_type(8)))  float  v8f;

// fp32 -> bf16 with round-to-nearest-even
__device__ __forceinline__ unsigned short f2bf(float f) {
    unsigned int u = __float_as_uint(f);
    unsigned int r = u + 0x7FFFu + ((u >> 16) & 1u);
    return (unsigned short)(r >> 16);
}

// ---------------- conversion kernels ----------------
__global__ void cvt_bf16_kernel(const float* __restrict__ src,
                                unsigned short* __restrict__ dst, int n) {
    int i = blockIdx.x * blockDim.x + threadIdx.x;
    if (i < n) dst[i] = f2bf(src[i]);
}

// src: [R, C] fp32 row-major  ->  dst: [C, R] bf16 row-major (transposed)
__global__ void tcvt_bf16_kernel(const float* __restrict__ src,
                                 unsigned short* __restrict__ dst, int R, int C) {
    int i = blockIdx.x * blockDim.x + threadIdx.x;
    if (i < R * C) {
        int r = i / C, c = i % C;
        dst[(size_t)c * R + r] = f2bf(src[i]);
    }
}

// ---------------- WMMA bf16 GEMM:  C[M,Ncols] = A[M,Kd] * Bt[Ncols,Kd]^T + bias ----------------
// A and Bt are bf16 row-major; both fragments load as two 16B contiguous chunks/lane.
// Register tile per wave: 32M x 32N (2 A-frags x 2 B-frags -> 4 WMMAs per 32-K step).
__device__ __forceinline__ v16bf load_frag(const unsigned short* __restrict__ rowp, int kb) {
    v8bf lo = *reinterpret_cast<const v8bf*>(rowp + kb);
    v8bf hi = *reinterpret_cast<const v8bf*>(rowp + kb + 16);
    v16bf f;
#pragma unroll
    for (int i = 0; i < 8; ++i) { f[i] = lo[i]; f[i + 8] = hi[i]; }
    return f;
}

__global__ void gemm_bf16_wmma_kernel(const unsigned short* __restrict__ A,
                                      const unsigned short* __restrict__ Bt,
                                      const float* __restrict__ bias,
                                      float* __restrict__ C,
                                      int Kd, int Ncols) {
    const int lane = threadIdx.x;            // wave32
    const int half = lane >> 4;              // 0: K lo-halves, 1: K hi-halves
    const int l    = lane & 15;
    const int tileN = blockIdx.x * 32;
    const int tileM = blockIdx.y * 32;

    const unsigned short* arow0 = A  + (size_t)(tileM + l) * Kd;
    const unsigned short* arow1 = A  + (size_t)(tileM + 16 + l) * Kd;
    const unsigned short* b0row = Bt + (size_t)(tileN + l) * Kd;
    const unsigned short* b1row = Bt + (size_t)(tileN + 16 + l) * Kd;

    v8f c00 = {};
    v8f c01 = {};
    v8f c10 = {};
    v8f c11 = {};

    for (int kk = 0; kk < Kd; kk += 32) {
        const int ka = kk + half * 8;
        // prefetch the next K-step (lowers to global_prefetch_b8)
        __builtin_prefetch(arow0 + kk + 32, 0, 1);
        __builtin_prefetch(arow1 + kk + 32, 0, 1);
        __builtin_prefetch(b0row + kk + 32, 0, 1);
        __builtin_prefetch(b1row + kk + 32, 0, 1);

        v16bf a0 = load_frag(arow0, ka);
        v16bf a1 = load_frag(arow1, ka);
        v16bf b0 = load_frag(b0row, ka);
        v16bf b1 = load_frag(b1row, ka);

        c00 = __builtin_amdgcn_wmma_f32_16x16x32_bf16(false, a0, false, b0,
                                                      (short)0, c00, false, false);
        c01 = __builtin_amdgcn_wmma_f32_16x16x32_bf16(false, a0, false, b1,
                                                      (short)0, c01, false, false);
        c10 = __builtin_amdgcn_wmma_f32_16x16x32_bf16(false, a1, false, b0,
                                                      (short)0, c10, false, false);
        c11 = __builtin_amdgcn_wmma_f32_16x16x32_bf16(false, a1, false, b1,
                                                      (short)0, c11, false, false);
    }

    const float bias0 = bias[tileN + l];
    const float bias1 = bias[tileN + 16 + l];
#pragma unroll
    for (int r = 0; r < 8; ++r) {
        const int row0 = tileM + half * 8 + r;
        const int row1 = tileM + 16 + half * 8 + r;
        C[(size_t)row0 * Ncols + tileN + l]      = c00[r] + bias0;
        C[(size_t)row0 * Ncols + tileN + 16 + l] = c01[r] + bias1;
        C[(size_t)row1 * Ncols + tileN + l]      = c10[r] + bias0;
        C[(size_t)row1 * Ncols + tileN + 16 + l] = c11[r] + bias1;
    }
}

// ---------------- RoPE (YaRN-style, matches reference) ----------------
__global__ void rope_kernel(float* __restrict__ qbuf, float* __restrict__ kbuf,
                            const int* __restrict__ positions) {
    const int t = blockIdx.x;      // token
    const int h = blockIdx.y;      // 0..N+K-1
    const int i = threadIdx.x;     // 0..31 rotary dim

    float* buf;
    size_t base;
    if (h < NHEADS) { buf = qbuf; base = ((size_t)t * NHEADS + h) * HDIM; }
    else            { buf = kbuf; base = ((size_t)t * KVHEADS + (h - NHEADS)) * HDIM; }

    const int pos = positions[t] % SEQ;
    const float fi = (float)i;
    const float TWO_PI = 6.283185307179586f;
    const float freq = __powf(150000.0f, (2.0f * fi) / 64.0f);
    const float concentration = 0.1f * logf(32.0f) + 1.0f;
    const float lg = logf(150000.0f);
    const float low  = 32.0f * logf(4096.0f / (32.0f * TWO_PI)) / lg;
    const float high = 32.0f * logf(4096.0f / TWO_PI) / lg;
    const float interp = 1.0f / (32.0f * freq);
    const float extrap = 1.0f / freq;
    float ramp = (fi - low) / (high - low);
    ramp = fminf(fmaxf(ramp, 0.0f), 1.0f);
    const float inv_freq = interp * ramp + extrap * (1.0f - ramp);

    const float tt = (float)pos * inv_freq;
    const float c = __cosf(tt) * concentration;
    const float s = __sinf(tt) * concentration;

    const float x1 = buf[base + i];
    const float x2 = buf[base + i + 32];
    buf[base + i]      = x1 * c - x2 * s;
    buf[base + i + 32] = x2 * c + x1 * s;
}

// ---------------- sliding-window attention with sinks ----------------
// one wave per (b, head, q); writes bf16 ctx[T, N*H] already divided by denom
__global__ void attn_kernel(const float* __restrict__ q,
                            const float* __restrict__ k,
                            const float* __restrict__ v,
                            const float* __restrict__ sinks,
                            unsigned short* __restrict__ ctx) {
    const int qpos = blockIdx.x;
    const int n    = blockIdx.y;
    const int b    = blockIdx.z;
    const int lane = threadIdx.x;          // 0..31
    const int g    = n >> 3;               // kv head (N/K = 8)
    const float sm_scale = 0.125f;         // 1/sqrt(64)

    __shared__ float qs[HDIM];
    __shared__ float ps[WINDOW];

    const float* qrow = q + ((size_t)(b * SEQ + qpos) * NHEADS + n) * HDIM;
    qs[lane] = qrow[lane];
    qs[lane + 32] = qrow[lane + 32];
    __syncthreads();

    int j0 = qpos - (WINDOW - 1);
    if (j0 < 0) j0 = 0;

    const float4* q4 = reinterpret_cast<const float4*>(qs);
    float sc[4];
#pragma unroll
    for (int r = 0; r < 4; ++r) {
        const int j = j0 + lane + 32 * r;
        if (j <= qpos) {
            const float* krow = k + ((size_t)(b * SEQ + j) * KVHEADS + g) * HDIM;
            const float4* k4 = reinterpret_cast<const float4*>(krow);
            float d = 0.0f;
#pragma unroll
            for (int t = 0; t < 16; ++t) {
                float4 kv = k4[t], qv = q4[t];
                d += kv.x * qv.x + kv.y * qv.y + kv.z * qv.z + kv.w * qv.w;
            }
            sc[r] = d * sm_scale;
        } else {
            sc[r] = NEG_INF;
        }
    }

    // wave max
    float m = fmaxf(fmaxf(sc[0], sc[1]), fmaxf(sc[2], sc[3]));
#pragma unroll
    for (int off = 16; off >= 1; off >>= 1) m = fmaxf(m, __shfl_xor(m, off));
    const float sk = sinks[n];
    m = fmaxf(m, sk);

    float psum = 0.0f;
#pragma unroll
    for (int r = 0; r < 4; ++r) {
        const float p = __expf(sc[r] - m);
        ps[lane + 32 * r] = p;
        psum += p;
    }
#pragma unroll
    for (int off = 16; off >= 1; off >>= 1) psum += __shfl_xor(psum, off);
    const float denom = psum + __expf(sk - m);
    __syncthreads();

    // each lane owns 2 output dims
    const int h0 = lane * 2;
    const int cnt = qpos - j0 + 1;
    float acc0 = 0.0f, acc1 = 0.0f;
    for (int jj = 0; jj < cnt; ++jj) {
        const float pp = ps[jj];
        const float* vrow = v + ((size_t)(b * SEQ + j0 + jj) * KVHEADS + g) * HDIM;
        acc0 += pp * vrow[h0];
        acc1 += pp * vrow[h0 + 1];
    }
    const float inv = 1.0f / denom;
    unsigned short* crow = ctx + ((size_t)(b * SEQ + qpos) * NHEADS + n) * HDIM;
    crow[h0]     = f2bf(acc0 * inv);
    crow[h0 + 1] = f2bf(acc1 * inv);
}

// ---------------- host launcher ----------------
extern "C" void kernel_launch(void* const* d_in, const int* in_sizes, int n_in,
                              void* d_out, int out_size, void* d_ws, size_t ws_size,
                              hipStream_t stream) {
    const float* x     = (const float*)d_in[0];
    const float* wq    = (const float*)d_in[1];
    const float* bq    = (const float*)d_in[2];
    const float* wk    = (const float*)d_in[3];
    const float* bk    = (const float*)d_in[4];
    const float* wv    = (const float*)d_in[5];
    const float* bv    = (const float*)d_in[6];
    const float* wo    = (const float*)d_in[7];
    const float* bo    = (const float*)d_in[8];
    const float* sinks = (const float*)d_in[9];
    const int* positions = (const int*)d_in[10];
    float* out = (float*)d_out;

    // workspace carving (aligned to 256B)
    char* p = (char*)d_ws;
    auto carve = [&](size_t bytes) {
        char* r = p;
        p += (bytes + 255) & ~(size_t)255;
        return r;
    };
    unsigned short* x_bf  = (unsigned short*)carve((size_t)TTOK * DMODEL * 2);
    unsigned short* wqT   = (unsigned short*)carve((size_t)NQ * DMODEL * 2);
    unsigned short* wkT   = (unsigned short*)carve((size_t)NKV * DMODEL * 2);
    unsigned short* wvT   = (unsigned short*)carve((size_t)NKV * DMODEL * 2);
    unsigned short* woT   = (unsigned short*)carve((size_t)DMODEL * NQ * 2);
    float*          qbuf  = (float*)carve((size_t)TTOK * NQ * 4);
    float*          kbuf  = (float*)carve((size_t)TTOK * NKV * 4);
    float*          vbuf  = (float*)carve((size_t)TTOK * NKV * 4);
    unsigned short* ctxbf = (unsigned short*)carve((size_t)TTOK * NQ * 2);

    // 1) convert x and transpose+convert weights to bf16
    {
        int n = TTOK * DMODEL;
        cvt_bf16_kernel<<<(n + 255) / 256, 256, 0, stream>>>(x, x_bf, n);
    }
    tcvt_bf16_kernel<<<((DMODEL * NQ) + 255) / 256, 256, 0, stream>>>(wq, wqT, DMODEL, NQ);
    tcvt_bf16_kernel<<<((DMODEL * NKV) + 255) / 256, 256, 0, stream>>>(wk, wkT, DMODEL, NKV);
    tcvt_bf16_kernel<<<((DMODEL * NKV) + 255) / 256, 256, 0, stream>>>(wv, wvT, DMODEL, NKV);
    tcvt_bf16_kernel<<<((NQ * DMODEL) + 255) / 256, 256, 0, stream>>>(wo, woT, NQ, DMODEL);

    // 2) QKV projections (WMMA bf16, fp32 accum), 32x32 tile per wave
    gemm_bf16_wmma_kernel<<<dim3(NQ / 32, TTOK / 32), 32, 0, stream>>>(
        x_bf, wqT, bq, qbuf, DMODEL, NQ);
    gemm_bf16_wmma_kernel<<<dim3(NKV / 32, TTOK / 32), 32, 0, stream>>>(
        x_bf, wkT, bk, kbuf, DMODEL, NKV);
    gemm_bf16_wmma_kernel<<<dim3(NKV / 32, TTOK / 32), 32, 0, stream>>>(
        x_bf, wvT, bv, vbuf, DMODEL, NKV);

    // 3) RoPE in place on q and k
    rope_kernel<<<dim3(TTOK, NHEADS + KVHEADS), 32, 0, stream>>>(qbuf, kbuf, positions);

    // 4) sliding-window attention with sinks -> bf16 ctx
    attn_kernel<<<dim3(SEQ, NHEADS, BATCH), 32, 0, stream>>>(qbuf, kbuf, vbuf, sinks, ctxbf);

    // 5) output projection (WMMA bf16, fp32 accum) -> d_out
    gemm_bf16_wmma_kernel<<<dim3(DMODEL / 32, TTOK / 32), 32, 0, stream>>>(
        ctxbf, woT, bo, out, NQ, DMODEL);
}
